// ContraFace_41102837023384
// MI455X (gfx1250) — matrix operand order repaired and checked by reading, not compiled, not run on hardware
//
#include <hip/hip_runtime.h>
#include <hip/hip_bf16.h>

typedef __attribute__((ext_vector_type(16))) _Float16 v16h;
typedef __attribute__((ext_vector_type(8)))  float    v8f;

#define BROWS 8192
#define DDIM  512
#define LSCALE 64.0f
#define EMA   0.99f

#define ROWS_PER_WG    128        // 8 waves * 16 rows
#define THREADS_PER_WG 256
#define COLS_PER_STEP  32
#define CSEG           8          // column segments (grid.y)
#define COLS_PER_SEG   (BROWS / CSEG)
#define LDS_K          (DDIM + 8) // pad 8 halves -> row stride 1040B = 4-bank shift

// ---------------------------------------------------------------------------
// Kernel 1: L2-normalize rows, emit fp16
// ---------------------------------------------------------------------------
__global__ __launch_bounds__(256) void cf_normalize_kernel(
    const float* __restrict__ in, _Float16* __restrict__ out)
{
  __shared__ float red[256];
  const int row = blockIdx.x;
  const int t = threadIdx.x;
  const float* p = in + (size_t)row * DDIM;
  float x0 = p[t];
  float x1 = p[t + 256];
  red[t] = x0 * x0 + x1 * x1;
  __syncthreads();
  for (int s = 128; s > 0; s >>= 1) {
    if (t < s) red[t] += red[t + s];
    __syncthreads();
  }
  float inv = rsqrtf(red[0]);
  _Float16* o = out + (size_t)row * DDIM;
  o[t]       = (_Float16)(x0 * inv);
  o[t + 256] = (_Float16)(x1 * inv);
}

// ---------------------------------------------------------------------------
// Kernel 2: fused cos-GEMM + per-row statistics (WMMA f16 -> f32)
//   grid.x : row tile (128 rows), grid.y : column segment (1024 cols)
//   Each wave owns 16 rows; A fragments for all K=512 held in registers.
//   B panels (32 cols x 512 K) staged into padded LDS with
//   GLOBAL_LOAD_ASYNC_TO_LDS_B128 (ASYNCcnt path, no VGPR round-trip).
// ---------------------------------------------------------------------------
__global__ __launch_bounds__(THREADS_PER_WG) void cf_cospass_kernel(
    const _Float16* __restrict__ f1n, const _Float16* __restrict__ f2n,
    const int* __restrict__ label,
    float* __restrict__ g_pos,           // [BROWS]
    float* __restrict__ g_neg_part,      // [CSEG][BROWS]
    float* __restrict__ g_exp_part)      // [CSEG][BROWS]
{
  __shared__ _Float16 bl[COLS_PER_STEP * LDS_K];   // 33,280 bytes

  const int tid    = threadIdx.x;
  const int wave   = tid >> 5;
  const int lane   = tid & 31;
  const int half16 = lane >> 4;      // 0: lanes 0-15, 1: lanes 16-31
  const int l16    = lane & 15;

  const int rbase = blockIdx.x * ROWS_PER_WG + wave * 16;
  const int cseg0 = blockIdx.y * COLS_PER_SEG;

  // ---- hoist A fragments (16 rows x K=512) into registers -----------------
  // ISA layout, 16-bit A 16x32: lanes 0-15 hold K {k0..k0+7, k0+16..k0+23},
  // lanes 16-31 hold K {k0+8..k0+15, k0+24..k0+31}, row M = lane&15.
  v16h a[16];
  {
    const _Float16* arow = f1n + (size_t)(rbase + l16) * DDIM;
    #pragma unroll
    for (int k = 0; k < 16; ++k) {
      const int k0 = k * 32;
      const float4* p0 = (const float4*)(arow + k0 + half16 * 8);
      const float4* p1 = (const float4*)(arow + k0 + 16 + half16 * 8);
      ((float4*)&a[k])[0] = *p0;
      ((float4*)&a[k])[1] = *p1;
    }
  }

  // Row labels for the 8 output rows this lane produces (M = half16*8 + v).
  int rlab[8];
  #pragma unroll
  for (int v = 0; v < 8; ++v) rlab[v] = label[rbase + half16 * 8 + v];

  float esum[8], emax[8];
  #pragma unroll
  for (int v = 0; v < 8; ++v) { esum[v] = 0.0f; emax[v] = 0.0f; }

  // B-panel staging: 8 threads per column row, 8 x 16B chunks per thread.
  const int ldr = tid >> 3;
  const int ldc = tid & 7;
  // Workgroup-relative LDS byte offset of this thread's staging row
  // (LDS aperture: flat addr[31:0] == LDS offset, HW adds LDS_BASE).
  const unsigned int lds_row = (unsigned int)(uintptr_t)(bl + ldr * LDS_K);

  for (int cbase = cseg0; cbase < cseg0 + COLS_PER_SEG; cbase += COLS_PER_STEP) {
    __syncthreads();  // previous panel fully consumed by all waves
    // ---- async stage of B panel (32 cols x 512 halves) into LDS ----------
    {
      const unsigned long long gsrc =
          (unsigned long long)(uintptr_t)(f2n + (size_t)(cbase + ldr) * DDIM);
      #pragma unroll
      for (int i = 0; i < 8; ++i) {
        const unsigned int       loff = lds_row + (unsigned)(ldc + 8 * i) * 16u;
        const unsigned long long goff = gsrc + (unsigned long long)(ldc + 8 * i) * 16ull;
        asm volatile("global_load_async_to_lds_b128 %0, %1, off"
                     :: "v"(loff), "v"(goff) : "memory");
      }
      // prefetch next panel toward L2 while this one lands (global_prefetch_b8)
      if (cbase + COLS_PER_STEP < cseg0 + COLS_PER_SEG)
        __builtin_prefetch((const void*)(f2n + (size_t)(cbase + COLS_PER_STEP + ldr) * DDIM + ldc * 64), 0, 1);
      asm volatile("s_wait_asynccnt 0x0" ::: "memory");
    }
    __syncthreads();

    // ---- two 16x16 sub-tiles per panel --------------------------------
    #pragma unroll
    for (int n0 = 0; n0 < COLS_PER_STEP; n0 += 16) {
      const int j = cbase + n0 + l16;            // this lane's output column
      const int clab = label[j];

      // B layout, 16-bit B 32x16: lane n holds column n, same K order as A
      // (lanes 0-15: K k0..k0+7, k0+16..k0+23; lanes 16-31: K k0+8..k0+15, k0+24..k0+31).
      const float4* bp = (const float4*)(bl + (n0 + l16) * LDS_K + half16 * 8);

      // Two interleaved accumulator chains + double-buffered B fragment.
      v8f c0 = {0.f, 0.f, 0.f, 0.f, 0.f, 0.f, 0.f, 0.f};
      v8f c1 = {0.f, 0.f, 0.f, 0.f, 0.f, 0.f, 0.f, 0.f};
      v16h bcur, bnxt;
      ((float4*)&bcur)[0] = bp[0];
      ((float4*)&bcur)[1] = bp[2];
      #pragma unroll
      for (int k = 0; k < 16; ++k) {
        if (k < 15) {                            // prefetch next fragment
          ((float4*)&bnxt)[0] = bp[(k + 1) * 4];
          ((float4*)&bnxt)[1] = bp[(k + 1) * 4 + 2];
        }
        if ((k & 1) == 0)
          c0 = __builtin_amdgcn_wmma_f32_16x16x32_f16(
              false, a[k], false, bcur, (short)0, c0, false, false);
        else
          c1 = __builtin_amdgcn_wmma_f32_16x16x32_f16(
              false, a[k], false, bcur, (short)0, c1, false, false);
        bcur = bnxt;
      }

      // ---- fused epilogue: clip, same-label mask, pos/neg/expsum -------
      #pragma unroll
      for (int v = 0; v < 8; ++v) {
        const int i = rbase + half16 * 8 + v;    // output row (C layout §7.12.2)
        float val = fminf(fmaxf(c0[v] + c1[v], -1.0f), 1.0f);
        if (i == j) {
          g_pos[i] = val;                        // true diagonal, written once
        } else {
          if (rlab[v] == clab) val = 0.0f;       // same-label off-diagonal -> 0
          esum[v] += __expf(LSCALE * val);       // masked zeros contribute exp(0)=1
          emax[v]  = fmaxf(emax[v], val);        // init 0 models diag->0 substitution
        }
      }
    }
  }

  // ---- reduce the 16 lanes sharing each output row (wave32 shuffles) ----
  #pragma unroll
  for (int v = 0; v < 8; ++v) {
    float s  = esum[v];
    float mx = emax[v];
    #pragma unroll
    for (int off = 1; off < 16; off <<= 1) {
      s += __shfl_xor(s, off, 32);
      mx = fmaxf(mx, __shfl_xor(mx, off, 32));
    }
    if (l16 == 0) {
      const int i = rbase + half16 * 8 + v;
      g_exp_part[(size_t)blockIdx.y * BROWS + i] = s;
      g_neg_part[(size_t)blockIdx.y * BROWS + i] = mx;
    }
  }
}

// ---------------------------------------------------------------------------
// Kernel 3: combine segments, compute m (EMA margin), then the loss
// ---------------------------------------------------------------------------
__global__ __launch_bounds__(1024) void cf_finalize_kernel(
    const float* __restrict__ g_pos,
    const float* __restrict__ g_neg_part,
    const float* __restrict__ g_exp_part,
    float* __restrict__ out)
{
  __shared__ float red[1024];
  __shared__ float m_sh;
  const int t = threadIdx.x;

  // phase 1: m = EMA * mean(pos - neg)
  float acc = 0.0f;
  for (int i = t; i < BROWS; i += 1024) {
    float ng = 0.0f;
    #pragma unroll
    for (int s = 0; s < CSEG; ++s)
      ng = fmaxf(ng, g_neg_part[(size_t)s * BROWS + i]);
    acc += g_pos[i] - ng;
  }
  red[t] = acc;
  __syncthreads();
  for (int s = 512; s > 0; s >>= 1) {
    if (t < s) red[t] += red[t + s];
    __syncthreads();
  }
  if (t == 0) m_sh = EMA * red[0] / (float)BROWS;
  __syncthreads();
  const float m = m_sh;

  // phase 2: loss = mean( log(expsum_i + exp(64*(pos_i-m))) - 64*(pos_i-m) )
  float lacc = 0.0f;
  for (int i = t; i < BROWS; i += 1024) {
    float es = 0.0f;
    #pragma unroll
    for (int s = 0; s < CSEG; ++s)
      es += g_exp_part[(size_t)s * BROWS + i];
    const float pd = LSCALE * (g_pos[i] - m);
    lacc += __logf(es + __expf(pd)) - pd;
  }
  red[t] = lacc;
  __syncthreads();
  for (int s = 512; s > 0; s >>= 1) {
    if (t < s) red[t] += red[t + s];
    __syncthreads();
  }
  if (t == 0) out[0] = red[0] / (float)BROWS;
}

// ---------------------------------------------------------------------------
extern "C" void kernel_launch(void* const* d_in, const int* in_sizes, int n_in,
                              void* d_out, int out_size, void* d_ws, size_t ws_size,
                              hipStream_t stream) {
  const float* f1    = (const float*)d_in[0];
  const float* f2    = (const float*)d_in[1];
  const int*   label = (const int*)d_in[2];
  float* out = (float*)d_out;

  char* ws = (char*)d_ws;
  const size_t featBytes = (size_t)BROWS * DDIM * sizeof(_Float16); // 8 MB each
  _Float16* f1n = (_Float16*)ws;
  _Float16* f2n = (_Float16*)(ws + featBytes);
  float* g_pos      = (float*)(ws + 2 * featBytes);
  float* g_neg_part = g_pos + BROWS;                 // [CSEG][BROWS]
  float* g_exp_part = g_neg_part + (size_t)CSEG * BROWS;

  cf_normalize_kernel<<<BROWS, 256, 0, stream>>>(f1, f1n);
  cf_normalize_kernel<<<BROWS, 256, 0, stream>>>(f2, f2n);

  dim3 grid(BROWS / ROWS_PER_WG, CSEG);
  cf_cospass_kernel<<<grid, THREADS_PER_WG, 0, stream>>>(
      f1n, f2n, label, g_pos, g_neg_part, g_exp_part);

  cf_finalize_kernel<<<1, 1024, 0, stream>>>(g_pos, g_neg_part, g_exp_part, out);
}
